// GATRatingPrediction_62259845922981
// MI455X (gfx1250) — matrix operand order repaired and checked by reading, not compiled, not run on hardware
//
#include <hip/hip_runtime.h>

// ---------------------------------------------------------------------------
// GAT rating prediction for MI455X (gfx1250, wave32, WMMA).
//   layer GEMMs  : v_wmma_f32_16x16x32_bf16, LDS-staged tiles + prefetch
//   segment softmax: 3-pass with order-preserving uint atomicMax
//   aggregation  : f32 scatter atomics (L2-resident on 192MB L2)
//   FC head      : reduced to per-node dots p1/p2 + edge gather
// ---------------------------------------------------------------------------

typedef __attribute__((ext_vector_type(16))) __bf16 v16bf;
typedef __attribute__((ext_vector_type(8)))  float  v8f;

union FragBF { v16bf h; unsigned u[8]; };

__device__ __forceinline__ unsigned pack_bf16x2(float a, float b) {
    unsigned ua = __float_as_uint(a), ub = __float_as_uint(b);
    ua += 0x7fffu + ((ua >> 16) & 1u);     // round-to-nearest-even
    ub += 0x7fffu + ((ub >> 16) & 1u);
    return (ua >> 16) | (ub & 0xffff0000u);
}

// order-preserving float<->uint for atomicMax over signed floats
__device__ __forceinline__ unsigned fenc(float f) {
    unsigned u = __float_as_uint(f);
    return (u & 0x80000000u) ? ~u : (u | 0x80000000u);
}
__device__ __forceinline__ float fdec(unsigned k) {
    unsigned u = (k & 0x80000000u) ? (k & 0x7fffffffu) : ~k;
    return __uint_as_float(u);
}

// ---------------------------------------------------------------------------
// GEMM: H[M,64] = X[M,K] * W[K,64]   (K = 256 or 64), bf16 WMMA, f32 accum.
// Block = 256 threads = 8 waves; each wave owns a 16x64 tile (4 C-frags).
// ---------------------------------------------------------------------------
template <int K>
__global__ __launch_bounds__(256)
void gat_gemm_wmma(const float* __restrict__ X, const float* __restrict__ W,
                   float* __restrict__ H, int M) {
    __shared__ unsigned Wl[(K / 2) * 64];   // bf16 pairs [kpair][n]  (B-frag order)
    __shared__ unsigned Xl[128 * 17];       // bf16 pairs [m][kpair], padded stride 17

    const int tid  = threadIdx.x;
    const int lane = tid & 31;
    const int wave = tid >> 5;
    const int half = lane >> 4;
    const int l16  = lane & 15;

    // stage W once (coalesced), converting f32 -> packed bf16 pairs
    for (int i = tid; i < (K / 2) * 64; i += 256) {
        int kp = i >> 6, n = i & 63;
        Wl[i] = pack_bf16x2(W[(2 * kp) * 64 + n], W[(2 * kp + 1) * 64 + n]);
    }

    const int rowBase0 = blockIdx.x * 128;
    v8f acc[4] = {};
    __syncthreads();

    for (int kb = 0; kb < K; kb += 32) {
        // stage 128 rows x 32 k of X (rows clamped; inactive waves still help)
        for (int i = tid; i < 128 * 16; i += 256) {
            int m = i >> 4, j = i & 15;
            int row = rowBase0 + m;
            row = row < M ? row : (M - 1);
            const float* p = X + (size_t)row * K + kb + 2 * j;
            Xl[m * 17 + j] = pack_bf16x2(p[0], p[1]);
            if (kb + 32 < K) __builtin_prefetch(p + 32, 0, 1);   // global_prefetch_b8
        }
        __syncthreads();

        // A fragment: lane -> M = l16, VGPR v -> K = 16*(v>>2) + 8*half + 2*(v&3)
        FragBF a;
        const int mloc = wave * 16 + l16;
#pragma unroll
        for (int v = 0; v < 8; ++v) {
            int j = ((v >> 2) << 3) + (half << 2) + (v & 3);   // kpair within kstep
            a.u[v] = Xl[mloc * 17 + j];
        }
#pragma unroll
        for (int jt = 0; jt < 4; ++jt) {
            // B fragment: lane -> N = jt*16 + l16, VGPR v -> K = 16*(v>>2) + 8*half + 2*(v&3)
            FragBF b;
            const int n = jt * 16 + l16;
            const int kpBase = (kb >> 1) + half * 4;
#pragma unroll
            for (int v = 0; v < 8; ++v) b.u[v] = Wl[(kpBase + ((v >> 2) << 3) + (v & 3)) * 64 + n];
            acc[jt] = __builtin_amdgcn_wmma_f32_16x16x32_bf16(
                false, a.h, false, b.h, (short)0, acc[jt], false, false);
        }
        __syncthreads();
    }

    // C/D layout: VGPR r -> M = r + 8*half, lane -> N = jt*16 + l16
    const int rowTile = (rowBase0 >> 4) + wave;
    const int mBase   = rowTile * 16;
    if (mBase + 16 <= M) {
        // wave-uniform fast path: straight-line coalesced stores, no exec dance
        float* __restrict__ out = H + (size_t)(mBase + 8 * half) * 64 + l16;
#pragma unroll
        for (int r = 0; r < 8; ++r)
#pragma unroll
            for (int jt = 0; jt < 4; ++jt)
                out[(size_t)r * 64 + jt * 16] = acc[jt][r];
    } else if (mBase < M) {
        // generic remainder (never taken when M % 16 == 0)
#pragma unroll
        for (int jt = 0; jt < 4; ++jt)
#pragma unroll
            for (int r = 0; r < 8; ++r) {
                int m = mBase + r + 8 * half;
                if (m < M) H[(size_t)m * 64 + jt * 16 + l16] = acc[jt][r];
            }
    }
}

// per-node double dot: oa[n] = H[n,:].va ; ob[n] = H[n,:].vb   (len 64)
__global__ void gat_node_dot2(const float* __restrict__ H,
                              const float* __restrict__ va, const float* __restrict__ vb,
                              float* __restrict__ oa, float* __restrict__ ob, int N) {
    int n = blockIdx.x * blockDim.x + threadIdx.x;
    if (n >= N) return;
    const float4* h = (const float4*)(H + (size_t)n * 64);
    const float4* A = (const float4*)va;
    const float4* B = (const float4*)vb;
    float sa = 0.f, sb = 0.f;
#pragma unroll
    for (int i = 0; i < 16; ++i) {
        float4 hv = h[i], av = A[i], bv = B[i];
        sa += hv.x * av.x + hv.y * av.y + hv.z * av.z + hv.w * av.w;
        sb += hv.x * bv.x + hv.y * bv.y + hv.z * bv.z + hv.w * bv.w;
    }
    oa[n] = sa;
    ob[n] = sb;
}

__global__ void gat_seg_init(float* __restrict__ agg, float* __restrict__ ssum,
                             unsigned* __restrict__ mkey, int N) {
    int i = blockIdx.x * blockDim.x + threadIdx.x;
    if (i < N * 64) agg[i] = 0.f;
    if (i < N) { ssum[i] = 0.f; mkey[i] = 0u; }   // 0 < fenc(x) for all finite x
}

// pass 1: e = leakyrelu(a_src[s]+a_dst[d]); segment max via uint atomicMax
__global__ void gat_edge_max(const int* __restrict__ src, const int* __restrict__ dst,
                             const float* __restrict__ as, const float* __restrict__ ad,
                             float* __restrict__ ebuf, unsigned* __restrict__ mkey,
                             int E, int N) {
    int i = blockIdx.x * blockDim.x + threadIdx.x;
    if (i >= E + N) return;
    int s = (i < E) ? src[i] : (i - E);
    int d = (i < E) ? dst[i] : (i - E);
    float e = as[s] + ad[d];
    e = (e >= 0.f) ? e : 0.2f * e;
    ebuf[i] = e;
    atomicMax(&mkey[d], fenc(e));
}

// pass 2: w = exp(e - m[d]); segment sum
__global__ void gat_edge_exp(const int* __restrict__ src, const int* __restrict__ dst,
                             float* __restrict__ ebuf, const unsigned* __restrict__ mkey,
                             float* __restrict__ ssum, int E, int N) {
    int i = blockIdx.x * blockDim.x + threadIdx.x;
    if (i >= E + N) return;
    int d = (i < E) ? dst[i] : (i - E);
    float w = __expf(ebuf[i] - fdec(mkey[d]));
    ebuf[i] = w;
    atomicAdd(&ssum[d], w);
}

// pass 3: agg[d,:] += (w/(s[d]+eps)) * h[s,:]  -- 16 threads/edge, float4 each
__global__ void gat_edge_agg(const int* __restrict__ src, const int* __restrict__ dst,
                             const float* __restrict__ wbuf, const float* __restrict__ ssum,
                             const float* __restrict__ H, float* __restrict__ agg,
                             int E, int N) {
    long long gid = (long long)blockIdx.x * blockDim.x + threadIdx.x;
    int i = (int)(gid >> 4);
    int t = (int)(gid & 15);
    if (i >= E + N) return;
    int s = (i < E) ? src[i] : (i - E);
    int d = (i < E) ? dst[i] : (i - E);
    float alpha = wbuf[i] / (ssum[d] + 1e-16f);
    float4 hv = *(const float4*)(H + (size_t)s * 64 + t * 4);
    float* o = agg + (size_t)d * 64 + t * 4;
    atomicAdd(o + 0, alpha * hv.x);
    atomicAdd(o + 1, alpha * hv.y);
    atomicAdd(o + 2, alpha * hv.z);
    atomicAdd(o + 3, alpha * hv.w);
}

__global__ void gat_bias_relu(float* __restrict__ agg, const float* __restrict__ b, int N) {
    int i = blockIdx.x * blockDim.x + threadIdx.x;
    if (i >= N * 64) return;
    float v = agg[i] + b[i & 63];
    agg[i] = v > 0.f ? v : 0.f;
}

// head: out[e] = p1[src_e] + p2[dst_e] + fc_b  (original edges only)
__global__ void gat_edge_pred(const int* __restrict__ src, const int* __restrict__ dst,
                              const float* __restrict__ p1, const float* __restrict__ p2,
                              const float* __restrict__ fcb, float* __restrict__ out, int E) {
    int i = blockIdx.x * blockDim.x + threadIdx.x;
    if (i >= E) return;
    out[i] = p1[src[i]] + p2[dst[i]] + fcb[0];
}

// ---------------------------------------------------------------------------
extern "C" void kernel_launch(void* const* d_in, const int* in_sizes, int n_in,
                              void* d_out, int out_size, void* d_ws, size_t ws_size,
                              hipStream_t stream) {
    const float* x    = (const float*)d_in[0];
    const int*   ei   = (const int*)d_in[1];
    const float* W1   = (const float*)d_in[2];
    const float* a1s  = (const float*)d_in[3];
    const float* a1d  = (const float*)d_in[4];
    const float* b1   = (const float*)d_in[5];
    const float* W2   = (const float*)d_in[6];
    const float* a2s  = (const float*)d_in[7];
    const float* a2d  = (const float*)d_in[8];
    const float* b2   = (const float*)d_in[9];
    const float* fcw  = (const float*)d_in[10];   // [128] : first 64 = src, next 64 = dst
    const float* fcb  = (const float*)d_in[11];

    const int N   = in_sizes[0] / 256;
    const int E   = in_sizes[1] / 2;
    const int tot = E + N;
    const int* src = ei;
    const int* dst = ei + E;

    // carve workspace (256B-aligned blocks)
    char* ws = (char*)d_ws;
    auto carve = [&](size_t bytes) {
        void* p = (void*)ws;
        ws += (bytes + 255) & ~(size_t)255;
        return p;
    };
    float*    h    = (float*)carve((size_t)N * 64 * 4);
    float*    agg  = (float*)carve((size_t)N * 64 * 4);
    float*    as   = (float*)carve((size_t)N * 4);
    float*    ad   = (float*)carve((size_t)N * 4);
    unsigned* mkey = (unsigned*)carve((size_t)N * 4);
    float*    ssum = (float*)carve((size_t)N * 4);
    float*    ebuf = (float*)carve((size_t)tot * 4);
    float*    p1   = (float*)carve((size_t)N * 4);
    float*    p2   = (float*)carve((size_t)N * 4);

    const int nb_node  = (N + 255) / 256;
    const int nb_nf    = (N * 64 + 255) / 256;
    const int nb_edge  = (tot + 255) / 256;
    const int nb_agg   = (int)(((long long)tot * 16 + 255) / 256);
    const int nb_gemm  = (N + 127) / 128;

    // ---- layer 1 ----
    gat_gemm_wmma<256><<<nb_gemm, 256, 0, stream>>>(x, W1, h, N);
    gat_node_dot2<<<nb_node, 256, 0, stream>>>(h, a1s, a1d, as, ad, N);
    gat_seg_init<<<nb_nf, 256, 0, stream>>>(agg, ssum, mkey, N);
    gat_edge_max<<<nb_edge, 256, 0, stream>>>(src, dst, as, ad, ebuf, mkey, E, N);
    gat_edge_exp<<<nb_edge, 256, 0, stream>>>(src, dst, ebuf, mkey, ssum, E, N);
    gat_edge_agg<<<nb_agg, 256, 0, stream>>>(src, dst, ebuf, ssum, h, agg, E, N);
    gat_bias_relu<<<nb_nf, 256, 0, stream>>>(agg, b1, N);      // agg = layer-2 input

    // ---- layer 2 ----
    gat_gemm_wmma<64><<<nb_gemm, 256, 0, stream>>>(agg, W2, h, N);   // h = h2
    gat_node_dot2<<<nb_node, 256, 0, stream>>>(h, a2s, a2d, as, ad, N);
    gat_seg_init<<<nb_nf, 256, 0, stream>>>(agg, ssum, mkey, N);     // agg reused
    gat_edge_max<<<nb_edge, 256, 0, stream>>>(src, dst, as, ad, ebuf, mkey, E, N);
    gat_edge_exp<<<nb_edge, 256, 0, stream>>>(src, dst, ebuf, mkey, ssum, E, N);
    gat_edge_agg<<<nb_agg, 256, 0, stream>>>(src, dst, ebuf, ssum, h, agg, E, N);
    gat_bias_relu<<<nb_nf, 256, 0, stream>>>(agg, b2, N);      // agg = final node feats

    // ---- head ----
    gat_node_dot2<<<nb_node, 256, 0, stream>>>(agg, fcw, fcw + 64, p1, p2, N);
    gat_edge_pred<<<(E + 255) / 256, 256, 0, stream>>>(src, dst, p1, p2, fcb,
                                                       (float*)d_out, E);
}
